// TransformerBlock_84524956385749
// MI455X (gfx1250) — matrix operand order repaired and checked by reading, not compiled, not run on hardware
//
#include <hip/hip_runtime.h>
#include <hip/hip_bf16.h>

#define D_MODEL 512
#define NUM_HEADS 8
#define HEAD_DIM 64
#define FFN_DIM 2048
#define BATCH 2
#define SEQ 4096
#define BT_ROWS (BATCH * SEQ)   // 8192

typedef _Float16 v16h __attribute__((ext_vector_type(16)));
typedef _Float16 v8h  __attribute__((ext_vector_type(8)));
typedef float    v8f  __attribute__((ext_vector_type(8)));

// ---------------------------------------------------------------------------
// WMMA helpers (CDNA5 wave32, V_WMMA_F32_16X16X32_F16)
// ---------------------------------------------------------------------------
__device__ __forceinline__ v8f wmma_f16(v16h a, v16h b, v8f c) {
  return __builtin_amdgcn_wmma_f32_16x16x32_f16(false, a, false, b, (short)0, c,
                                                false, false);
}

// A-fragment, 16x32 f16, row-major source, leading dim `ld` (halves).
// lanes 0-15: row=lane, K 0..15 ; lanes 16-31: row=lane-16, K 16..31
__device__ __forceinline__ v16h load_frag_a(const _Float16* base, int ld, int lane) {
  int row = lane & 15;
  int kb  = (lane >> 4) * 16;
  const _Float16* p = base + row * ld + kb;
  v8h lo = *(const v8h*)(p);
  v8h hi = *(const v8h*)(p + 8);
  return __builtin_shufflevector(lo, hi, 0, 1, 2, 3, 4, 5, 6, 7,
                                 8, 9, 10, 11, 12, 13, 14, 15);
}

// B-fragment, 32x16 f16 from B^T stored row-major [N][K]: lane = column n,
// K-run (lane/16)*16..+15 => one contiguous 32-byte run per lane.
__device__ __forceinline__ v16h load_frag_b(const _Float16* baseT, int ld, int lane) {
  int col = lane & 15;
  int kb  = (lane >> 4) * 16;
  return *(const v16h*)(baseT + col * ld + kb);
}

// Row reductions within the 16-lane half-groups of a wave32 C-layout.
__device__ __forceinline__ float rowmax16(float v) {
#pragma unroll
  for (int m = 8; m >= 1; m >>= 1) v = fmaxf(v, __shfl_xor(v, m, 32));
  return v;
}
__device__ __forceinline__ float rowsum16(float v) {
#pragma unroll
  for (int m = 8; m >= 1; m >>= 1) v += __shfl_xor(v, m, 32);
  return v;
}

// Async global->LDS copy of 16 bytes (CDNA5 GLOBAL_LOAD_ASYNC_TO_LDS_B128,
// tracked with ASYNCcnt, no VGPR round trip).  ldsoff is the byte offset of
// the destination inside the block's static LDS allocation (offset 0 based).
__device__ __forceinline__ void async_copy_b128(unsigned ldsoff, const void* gptr) {
  asm volatile("global_load_async_to_lds_b128 %0, %1, off"
               :
               : "v"(ldsoff), "v"((unsigned long long)(uintptr_t)gptr)
               : "memory");
}
__device__ __forceinline__ void wait_async0() {
  asm volatile("s_wait_asynccnt 0x0" ::: "memory");
}
__device__ __forceinline__ void wait_ds0() {
  asm volatile("s_wait_dscnt 0x0" ::: "memory");
}

// ---------------------------------------------------------------------------
// Weight prep: fp32 [K][N] -> f16 transposed [N][K]  (2-D grid, no int div)
// ---------------------------------------------------------------------------
__global__ void transpose_to_f16(const float* __restrict__ in,
                                 _Float16* __restrict__ out, int K, int N) {
  int n = blockIdx.x * 256 + threadIdx.x;
  int k = blockIdx.y;
  if (n < N) out[(size_t)n * K + k] = (_Float16)in[(size_t)k * N + n];
}

// ---------------------------------------------------------------------------
// LayerNorm over D_MODEL=512, one block per row, fp32 in -> f16 out
// ---------------------------------------------------------------------------
__global__ void layernorm_f16(const float* __restrict__ x,
                              const float* __restrict__ g,
                              const float* __restrict__ b,
                              _Float16* __restrict__ out) {
  __shared__ float s1[256], s2[256];
  int row = blockIdx.x, t = threadIdx.x;
  const float* xr = x + row * D_MODEL;
  float v0 = xr[t], v1 = xr[t + 256];
  s1[t] = v0 + v1;
  s2[t] = v0 * v0 + v1 * v1;
  __syncthreads();
  for (int off = 128; off > 0; off >>= 1) {
    if (t < off) { s1[t] += s1[t + off]; s2[t] += s2[t + off]; }
    __syncthreads();
  }
  float mu = s1[0] * (1.0f / D_MODEL);
  float var = s2[0] * (1.0f / D_MODEL) - mu * mu;
  float rs = rsqrtf(var + 1e-5f);
  out[row * D_MODEL + t]       = (_Float16)((v0 - mu) * rs * g[t] + b[t]);
  out[row * D_MODEL + t + 256] = (_Float16)((v1 - mu) * rs * g[t + 256] + b[t + 256]);
}

// ---------------------------------------------------------------------------
// WMMA GEMM: C[M,N] = A[M,K] f16 * B^T[N,K] f16 + bias.
// Block tile 128x128, 8 waves (4 in M x 2 in N), each wave 32x64 = 2x4 C tiles.
// mode 1: scatter into q[B,H,T,64], k[B,H,T,64], vT[B,H,64,T] (f16)
// mode 2: outF = C + bias + resid (fp32)
// mode 3: outH = relu(C + bias)   (f16)
// ---------------------------------------------------------------------------
__global__ void gemm_wmma_f16(const _Float16* __restrict__ A,
                              const _Float16* __restrict__ BT,
                              const float* __restrict__ bias,
                              const float* __restrict__ resid,
                              float* __restrict__ outF,
                              _Float16* __restrict__ outH,
                              int M, int N, int K, int mode,
                              _Float16* __restrict__ qo,
                              _Float16* __restrict__ ko,
                              _Float16* __restrict__ vTo) {
  int tid = threadIdx.x, wid = tid >> 5, lane = tid & 31;
  int wm0 = blockIdx.y * 128 + (wid >> 1) * 32;
  int wn0 = blockIdx.x * 128 + (wid & 1) * 64;

  v8f c[2][4] = {};
#pragma unroll 2
  for (int kk = 0; kk < K; kk += 32) {
    v16h a0 = load_frag_a(A + (size_t)wm0 * K + kk, K, lane);
    v16h a1 = load_frag_a(A + (size_t)(wm0 + 16) * K + kk, K, lane);
    v16h bf[4];
#pragma unroll
    for (int j = 0; j < 4; j++)
      bf[j] = load_frag_b(BT + (size_t)(wn0 + 16 * j) * K + kk, K, lane);
#pragma unroll
    for (int j = 0; j < 4; j++) {
      c[0][j] = wmma_f16(a0, bf[j], c[0][j]);
      c[1][j] = wmma_f16(a1, bf[j], c[1][j]);
    }
  }

  int g = lane >> 4, nl = lane & 15;
#pragma unroll
  for (int i = 0; i < 2; i++) {
#pragma unroll
    for (int j = 0; j < 4; j++) {
#pragma unroll
      for (int e = 0; e < 8; e++) {
        int m = wm0 + 16 * i + e + 8 * g;   // C-layout: VGPR e -> row e (+8 hi half)
        int n = wn0 + 16 * j + nl;          // lane%16 -> column
        float val = c[i][j][e] + bias[n];
        if (mode == 1) {
          int which = n >> 9;               // / 512
          int cc = n & 511;
          int h = cc >> 6, d = cc & 63;
          int bb = m >> 12, t = m & 4095;
          if (which == 0) {
            qo[(size_t)(((bb * NUM_HEADS + h) * SEQ + t)) * HEAD_DIM + d] = (_Float16)val;
          } else if (which == 1) {
            ko[(size_t)(((bb * NUM_HEADS + h) * SEQ + t)) * HEAD_DIM + d] = (_Float16)val;
          } else {
            vTo[(size_t)(((bb * NUM_HEADS + h) * HEAD_DIM + d)) * SEQ + t] = (_Float16)val;
          }
        } else if (mode == 2) {
          outF[(size_t)m * N + n] = val + resid[(size_t)m * N + n];
        } else {  // relu
          outH[(size_t)m * N + n] = (_Float16)fmaxf(val, 0.0f);
        }
      }
    }
  }
}

// ---------------------------------------------------------------------------
// Flash attention.  Block = 8 waves = one (b,h) x 128 Q rows; each wave owns a
// 16-row Q tile.  Per 32-key step, the K tile (32x64) and V^T tile (64x32) are
// staged into LDS with GLOBAL_LOAD_ASYNC_TO_LDS_B128 and DOUBLE-BUFFERED so
// the async DMA of tile i+1 overlaps the WMMA/softmax of tile i.
//
// LDS layout (static shared @ LDS offset 0), 24 KiB total:
//   [0,     8192)  buffer 0: K tile 32x64 halves | V^T tile 64x32 halves
//   [8192, 16384)  buffer 1: same
//   [16384,24576)  pbuf: 8 waves x 16x32 halves (P tile per wave)
// ---------------------------------------------------------------------------
__global__ void attn_flash(const _Float16* __restrict__ Q,
                           const _Float16* __restrict__ Kt,
                           const _Float16* __restrict__ Vt,
                           _Float16* __restrict__ O) {
  __shared__ _Float16 smem[12288];   // 24 KiB
  int tid = threadIdx.x, wid = tid >> 5, lane = tid & 31;
  _Float16* pb = smem + 8192 + wid * 512;      // [16][32] per wave

  int qt = blockIdx.x & 31;                    // T/128 = 32 q-panels
  int hb = blockIdx.x >> 5;
  int h = hb & (NUM_HEADS - 1);
  int b = hb >> 3;

  const _Float16* qbase =
      Q + ((size_t)((b * NUM_HEADS + h) * SEQ + qt * 128 + wid * 16)) * HEAD_DIM;
  const _Float16* kbase = Kt + ((size_t)((b * NUM_HEADS + h) * SEQ)) * HEAD_DIM;
  const _Float16* vbase = Vt + ((size_t)((b * NUM_HEADS + h) * HEAD_DIM)) * SEQ;

  int g = lane >> 4, nl = lane & 15;
  v16h aq0 = load_frag_a(qbase, HEAD_DIM, lane);       // d 0..31
  v16h aq1 = load_frag_a(qbase + 32, HEAD_DIM, lane);  // d 32..63

  v8f o0 = {}, o1 = {}, o2 = {}, o3 = {};
  float mrow[8], lrow[8];
#pragma unroll
  for (int e = 0; e < 8; e++) { mrow[e] = -1e30f; lrow[e] = 0.0f; }
  const float scale = 0.125f;  // 1/sqrt(64)

  // Per-thread staging: each of 256 threads copies one 16B chunk of the K
  // tile and one of the V tile per stage.
  int krow = tid >> 3, kch = tid & 7;   // K: 32 rows x 8 chunks of 16B
  int vrow = tid >> 2, vch = tid & 3;   // V: 64 rows x 4 chunks of 16B

  auto stage = [&](int kt, int s) {
    unsigned base = (unsigned)(s * 8192);
    async_copy_b128(base + (unsigned)(tid * 16),
                    kbase + (size_t)(kt + krow) * HEAD_DIM + kch * 8);
    async_copy_b128(base + 4096u + (unsigned)(tid * 16),
                    vbase + (size_t)vrow * SEQ + kt + vch * 8);
  };

  stage(0, 0);
  for (int it = 0; it < SEQ / 32; ++it) {
    int kt = it * 32;
    int s = it & 1;
    wait_async0();       // buffer s ready (this wave's copies)
    __syncthreads();     // ... and everyone else's; also: all waves finished
                         // reading buffer s two iterations ago
    if (it + 1 < SEQ / 32) stage(kt + 32, s ^ 1);   // overlap DMA with compute

    const _Float16* kbuf = smem + s * 4096;          // [32][64]
    const _Float16* vbuf = smem + s * 4096 + 2048;   // [64][32]

    // ---- S = Q K^T, two 16-key subtiles, fragments read from LDS ----
    v8f s0 = {}, s1 = {};
    {
      v16h bk0 = load_frag_b(kbuf, HEAD_DIM, lane);
      v16h bk1 = load_frag_b(kbuf + 32, HEAD_DIM, lane);
      s0 = wmma_f16(aq0, bk0, s0);
      s0 = wmma_f16(aq1, bk1, s0);
      v16h bk2 = load_frag_b(kbuf + 16 * HEAD_DIM, HEAD_DIM, lane);
      v16h bk3 = load_frag_b(kbuf + 16 * HEAD_DIM + 32, HEAD_DIM, lane);
      s1 = wmma_f16(aq0, bk2, s1);
      s1 = wmma_f16(aq1, bk3, s1);
    }
    // ---- online softmax (joint over both subtiles) ----
#pragma unroll
    for (int e = 0; e < 8; e++) {
      float x0 = s0[e] * scale, x1 = s1[e] * scale;
      float tmax = rowmax16(fmaxf(x0, x1));
      float mn = fmaxf(mrow[e], tmax);
      float al = __expf(mrow[e] - mn);
      float p0 = __expf(x0 - mn);
      float p1 = __expf(x1 - mn);
      float rs = rowsum16(p0 + p1);
      lrow[e] = lrow[e] * al + rs;
      mrow[e] = mn;
      int r = e + 8 * g;                     // C-layout row
      pb[r * 32 + nl] = (_Float16)p0;
      pb[r * 32 + 16 + nl] = (_Float16)p1;
      o0[e] *= al; o1[e] *= al; o2[e] *= al; o3[e] *= al;
    }
    wait_ds0();
    // ---- O += P * V ----
    v16h ap = load_frag_a(pb, 32, lane);
    o0 = wmma_f16(ap, load_frag_b(vbuf + 0 * 16 * 32, 32, lane), o0);
    o1 = wmma_f16(ap, load_frag_b(vbuf + 1 * 16 * 32, 32, lane), o1);
    o2 = wmma_f16(ap, load_frag_b(vbuf + 2 * 16 * 32, 32, lane), o2);
    o3 = wmma_f16(ap, load_frag_b(vbuf + 3 * 16 * 32, 32, lane), o3);
  }

  // ---- normalize and write out [B*T, 512] at col h*64+d ----
#pragma unroll
  for (int e = 0; e < 8; e++) {
    float inv = 1.0f / lrow[e];
    int row = b * SEQ + qt * 128 + wid * 16 + e + 8 * g;
    _Float16* op = O + (size_t)row * D_MODEL + h * HEAD_DIM + nl;
    op[0]  = (_Float16)(o0[e] * inv);
    op[16] = (_Float16)(o1[e] * inv);
    op[32] = (_Float16)(o2[e] * inv);
    op[48] = (_Float16)(o3[e] * inv);
  }
}

// ---------------------------------------------------------------------------
// Host-side orchestration
// ---------------------------------------------------------------------------
extern "C" void kernel_launch(void* const* d_in, const int* in_sizes, int n_in,
                              void* d_out, int out_size, void* d_ws, size_t ws_size,
                              hipStream_t stream) {
  const float* x      = (const float*)d_in[0];
  const float* ln1_g  = (const float*)d_in[1];
  const float* ln1_b  = (const float*)d_in[2];
  const float* w_qkv  = (const float*)d_in[3];
  const float* b_qkv  = (const float*)d_in[4];
  const float* w_out  = (const float*)d_in[5];
  const float* b_out  = (const float*)d_in[6];
  const float* ln2_g  = (const float*)d_in[7];
  const float* ln2_b  = (const float*)d_in[8];
  const float* w_ffn1 = (const float*)d_in[9];
  const float* b_ffn1 = (const float*)d_in[10];
  const float* w_ffn2 = (const float*)d_in[11];
  const float* b_ffn2 = (const float*)d_in[12];
  float* out = (float*)d_out;
  char* ws = (char*)d_ws;

  const size_t OFF_H16    = 0;
  const size_t OFF_WQKVT  = 8388608;
  const size_t OFF_WOUTT  = 9961472;
  const size_t OFF_WFFN1T = 10485760;
  const size_t OFF_WFFN2T = 12582912;
  const size_t OFF_Q      = 14680064;
  const size_t OFF_K      = 23068672;
  const size_t OFF_VT     = 31457280;
  const size_t OFF_ATTN   = 39845888;
  const size_t OFF_X2     = 48234496;
  const size_t OFF_H2     = 65011712;
  const size_t OFF_FFN1   = 73400320;

  _Float16* h16    = (_Float16*)(ws + OFF_H16);
  _Float16* wqkvT  = (_Float16*)(ws + OFF_WQKVT);
  _Float16* woutT  = (_Float16*)(ws + OFF_WOUTT);
  _Float16* wffn1T = (_Float16*)(ws + OFF_WFFN1T);
  _Float16* wffn2T = (_Float16*)(ws + OFF_WFFN2T);
  _Float16* q16    = (_Float16*)(ws + OFF_Q);
  _Float16* k16    = (_Float16*)(ws + OFF_K);
  _Float16* vT16   = (_Float16*)(ws + OFF_VT);
  _Float16* attn16 = (_Float16*)(ws + OFF_ATTN);
  float*    x2     = (float*)(ws + OFF_X2);
  _Float16* h2_16  = (_Float16*)(ws + OFF_H2);
  _Float16* ffn1o  = (_Float16*)(ws + OFF_FFN1);

  transpose_to_f16<<<dim3((3 * D_MODEL + 255) / 256, D_MODEL), 256, 0, stream>>>(
      w_qkv, wqkvT, D_MODEL, 3 * D_MODEL);
  transpose_to_f16<<<dim3((D_MODEL + 255) / 256, D_MODEL), 256, 0, stream>>>(
      w_out, woutT, D_MODEL, D_MODEL);
  transpose_to_f16<<<dim3((FFN_DIM + 255) / 256, D_MODEL), 256, 0, stream>>>(
      w_ffn1, wffn1T, D_MODEL, FFN_DIM);
  transpose_to_f16<<<dim3((D_MODEL + 255) / 256, FFN_DIM), 256, 0, stream>>>(
      w_ffn2, wffn2T, FFN_DIM, D_MODEL);

  layernorm_f16<<<BT_ROWS, 256, 0, stream>>>(x, ln1_g, ln1_b, h16);

  gemm_wmma_f16<<<dim3(3 * D_MODEL / 128, BT_ROWS / 128), 256, 0, stream>>>(
      h16, wqkvT, b_qkv, nullptr, nullptr, nullptr,
      BT_ROWS, 3 * D_MODEL, D_MODEL, 1, q16, k16, vT16);

  attn_flash<<<BATCH * NUM_HEADS * (SEQ / 128), 256, 0, stream>>>(
      q16, k16, vT16, attn16);

  gemm_wmma_f16<<<dim3(D_MODEL / 128, BT_ROWS / 128), 256, 0, stream>>>(
      attn16, woutT, b_out, x, x2, nullptr,
      BT_ROWS, D_MODEL, D_MODEL, 2, nullptr, nullptr, nullptr);

  layernorm_f16<<<BT_ROWS, 256, 0, stream>>>(x2, ln2_g, ln2_b, h2_16);

  gemm_wmma_f16<<<dim3(FFN_DIM / 128, BT_ROWS / 128), 256, 0, stream>>>(
      h2_16, wffn1T, b_ffn1, nullptr, nullptr, ffn1o,
      BT_ROWS, FFN_DIM, D_MODEL, 3, nullptr, nullptr, nullptr);

  gemm_wmma_f16<<<dim3(D_MODEL / 128, BT_ROWS / 128), 256, 0, stream>>>(
      ffn1o, wffn2T, b_ffn2, x2, out, nullptr,
      BT_ROWS, D_MODEL, FFN_DIM, 2, nullptr, nullptr, nullptr);

  (void)in_sizes; (void)n_in; (void)out_size; (void)ws_size;
}
